// Sup_CCL_Loss_27453430956191
// MI455X (gfx1250) — matrix operand order repaired and checked by reading, not compiled, not run on hardware
//
#include <hip/hip_runtime.h>
#include <hip/hip_bf16.h>

#define B_SZ 4096
#define P_SZ 12
#define DIN 640
#define DM 128
#define MT 128             // rows per workgroup
#define NMT (MT / 16)      // 8 m-tiles per wave
#define KS1 (DIN / 32)     // 20 k-steps, pass 1
#define KS2 (DM / 32)      // 4  k-steps, pass 2
#define XSTRIDE 40         // padded LDS stride (bf16) for x tile (mult of 8)
#define HSTRIDE 136        // padded LDS stride (bf16) for h tile (mult of 8)

typedef __attribute__((ext_vector_type(16))) __bf16 v16bf;
typedef __attribute__((ext_vector_type(8)))  float  v8f;

union Frag16 {
  v16bf v;
  uint4 q[2];
  unsigned short s[16];
};

__device__ __forceinline__ unsigned short f2bf_rne(float f) {
  unsigned u = __float_as_uint(f);
  u = (u + 0x7FFFu + ((u >> 16) & 1u)) >> 16;   // round-to-nearest-even
  return (unsigned short)u;
}

__device__ __forceinline__ unsigned short f2bf_trunc(float f) {
  return (unsigned short)(__float_as_uint(f) >> 16);   // 1 VALU op
}

// pack two floats -> one dword of 2 bf16 (lo in [15:0]); 1 VALU op via v_perm
__device__ __forceinline__ unsigned pack2bf(float lo, float hi) {
#if __has_builtin(__builtin_amdgcn_cvt_pk_bf16_f32)
  typedef __attribute__((ext_vector_type(2))) __bf16 v2bf;
  v2bf t = __builtin_amdgcn_cvt_pk_bf16_f32(lo, hi);
  return __builtin_bit_cast(unsigned, t);
#else
  // dst bytes = {hi.b3, hi.b2, lo.b3, lo.b2}  -> v_perm_b32
  return __builtin_amdgcn_perm(__float_as_uint(hi), __float_as_uint(lo),
                               0x07060302u);
#endif
}

// ---------------------------------------------------------------------------
// Pack fp32 weights [N=128, K] into bf16 WMMA-B fragment order:
// out[((nt*nks + ks)*32 + lane)*16 + e]  with  n = nt*16 + (lane&15),
// k = ks*32 + (lane>>4)*8 + (e&7) + (e>>3)*16   (B: lanes 0-15 hold K 0-7,16-23; 16-31 hold K 8-15,24-31)
// ---------------------------------------------------------------------------
__global__ void pack_w_kernel(const float* __restrict__ w,
                              unsigned short* __restrict__ out, int K) {
  int nks = K >> 5;
  int total = 8 * nks * 32 * 16;
  int i = blockIdx.x * blockDim.x + threadIdx.x;
  if (i >= total) return;
  int e    = i & 15;
  int lane = (i >> 4) & 31;
  int rest = i >> 9;
  int ks   = rest % nks;
  int nt   = rest / nks;
  int n = nt * 16 + (lane & 15);
  int k = ks * 32 + ((lane >> 4) << 3) + (e & 7) + ((e >> 3) << 4);
  out[i] = f2bf_rne(w[(size_t)n * K + k]);   // one-time: keep RNE for weights
}

// ---------------------------------------------------------------------------
// Fused embed: y = L2norm( relu(x @ w1^T + b1) @ w2^T + b2 ), x:[M,640], y:[M,128]
// 128 rows per WG, 8 waves, wave w owns output columns [16w, 16w+16).
// x-tile converted fp32->bf16 once at staging (v_perm), double-buffered;
// B-fragments pipelined across the barrier; A-fragments pipelined one deep.
// ---------------------------------------------------------------------------
__launch_bounds__(256, 2)
__global__ void embed_kernel(const float* __restrict__ x,
                             const unsigned short* __restrict__ w1p,
                             const float* __restrict__ b1,
                             const unsigned short* __restrict__ w2p,
                             const float* __restrict__ b2,
                             float* __restrict__ out) {
  __shared__ __align__(16) unsigned short xsb[2][MT * XSTRIDE];
  __shared__ __align__(16) unsigned short hs[MT * HSTRIDE];
  __shared__ float norms[MT];

  const int wave = threadIdx.x >> 5;   // n-tile 0..7
  const int lane = threadIdx.x & 31;
  const int hi   = lane >> 4;
  const int ln   = lane & 15;
  const size_t m0 = (size_t)blockIdx.x * MT;
  const int kb1  = hi * 8;             // A layout: lane group selects K-subblock

  // stage one 128x32 tile: 16 fp32/thread -> 2x 16B bf16 stores
  const int sr  = threadIdx.x >> 2;          // row 0..63 (+64 second half)
  const int sc8 = (threadIdx.x & 3) << 3;    // col 0,8,16,24
  auto stage = [&](int ks, int buf) {
#pragma unroll
    for (int h = 0; h < 2; ++h) {
      const int r = sr + h * 64;
      const float4* src = (const float4*)&x[(m0 + r) * DIN + ks * 32 + sc8];
      float4 f0 = src[0];
      float4 f1 = src[1];
      uint4 d;
      d.x = pack2bf(f0.x, f0.y);
      d.y = pack2bf(f0.z, f0.w);
      d.z = pack2bf(f1.x, f1.y);
      d.w = pack2bf(f1.z, f1.w);
      *(uint4*)&xsb[buf][r * XSTRIDE + sc8] = d;
    }
  };
  auto loadB1 = [&](Frag16& f, int ks) {
    const uint4* bp =
        (const uint4*)&w1p[(size_t)((wave * KS1 + ks) * 32 + lane) * 16];
    f.q[0] = bp[0];
    f.q[1] = bp[1];
  };
  auto loadA1 = [&](Frag16& f, int mt, int buf) {
    const unsigned short* row = &xsb[buf][(mt * 16 + ln) * XSTRIDE];
    f.q[0] = *(const uint4*)&row[kb1];        // K kb..kb+7
    f.q[1] = *(const uint4*)&row[kb1 + 16];   // K kb+16..kb+23
  };

  const v8f zero8 = {0.f, 0.f, 0.f, 0.f, 0.f, 0.f, 0.f, 0.f};
  v8f acc[NMT];
#pragma unroll
  for (int mt = 0; mt < NMT; ++mt) acc[mt] = zero8;

  // ---- Pass 1: h = relu(x @ w1^T + b1), K = 640 ----
  Frag16 bfr[2];
  loadB1(bfr[0], 0);
  stage(0, 0);
  __syncthreads();

#pragma unroll 2
  for (int ks = 0; ks < KS1; ++ks) {
    const int cur = ks & 1;
    if (ks + 1 < KS1) {
      loadB1(bfr[cur ^ 1], ks + 1);      // issue next B before this barrier
      stage(ks + 1, cur ^ 1);            // overlap staging with WMMAs
    }
    if (ks + 2 < KS1)                    // gfx1250 global_prefetch_b8
      __builtin_prefetch(&x[(m0 + sr) * DIN + (ks + 2) * 32 + sc8], 0, 1);

    Frag16 af[2];
    loadA1(af[0], 0, cur);
#pragma unroll
    for (int mt = 0; mt < NMT; ++mt) {
      if (mt + 1 < NMT) loadA1(af[(mt + 1) & 1], mt + 1, cur);  // pipeline A
      acc[mt] = __builtin_amdgcn_wmma_f32_16x16x32_bf16(
          false, af[mt & 1].v, false, bfr[cur].v, (short)0, acc[mt], false,
          false);
    }
    __syncthreads();
  }

  // bias + relu -> bf16 h tile in LDS
  const float bias1 = b1[wave * 16 + ln];
#pragma unroll
  for (int mt = 0; mt < NMT; ++mt) {
#pragma unroll
    for (int j = 0; j < 8; ++j) {
      float v = acc[mt][j] + bias1;
      v = v > 0.f ? v : 0.f;
      int r = mt * 16 + hi * 8 + j;      // C layout: row = j + 8*(lane/16)
      hs[r * HSTRIDE + wave * 16 + ln] = f2bf_trunc(v);
    }
  }
  if (threadIdx.x < MT) norms[threadIdx.x] = 0.f;
  __syncthreads();

  // ---- Pass 2: y = h @ w2^T + b2, K = 128 ----
  v8f acc2[NMT];
#pragma unroll
  for (int mt = 0; mt < NMT; ++mt) acc2[mt] = zero8;

#pragma unroll
  for (int ks = 0; ks < KS2; ++ks) {
    Frag16 bf;
    const uint4* bp =
        (const uint4*)&w2p[(size_t)((wave * KS2 + ks) * 32 + lane) * 16];
    bf.q[0] = bp[0];
    bf.q[1] = bp[1];
    const int kb = ks * 32 + hi * 8;
    auto loadA2 = [&](Frag16& f, int mt) {
      const unsigned short* hrow = &hs[(mt * 16 + ln) * HSTRIDE];
      f.q[0] = *(const uint4*)&hrow[kb];
      f.q[1] = *(const uint4*)&hrow[kb + 16];
    };
    Frag16 af[2];
    loadA2(af[0], 0);
#pragma unroll
    for (int mt = 0; mt < NMT; ++mt) {
      if (mt + 1 < NMT) loadA2(af[(mt + 1) & 1], mt + 1);       // pipeline A
      acc2[mt] = __builtin_amdgcn_wmma_f32_16x16x32_bf16(
          false, af[mt & 1].v, false, bf.v, (short)0, acc2[mt], false, false);
    }
  }

  // bias + row-wise sum(y^2) via LDS float atomics
  const float bias2 = b2[wave * 16 + ln];
#pragma unroll
  for (int mt = 0; mt < NMT; ++mt) {
#pragma unroll
    for (int j = 0; j < 8; ++j) {
      float v = acc2[mt][j] + bias2;
      acc2[mt][j] = v;
      atomicAdd(&norms[mt * 16 + hi * 8 + j], v * v);
    }
  }
  __syncthreads();

  // normalize and store
#pragma unroll
  for (int mt = 0; mt < NMT; ++mt) {
#pragma unroll
    for (int j = 0; j < 8; ++j) {
      int r = mt * 16 + hi * 8 + j;
      float inv = rsqrtf(norms[r]);
      out[(m0 + r) * DM + wave * 16 + ln] = acc2[mt][j] * inv;
    }
  }
}

// ---------------------------------------------------------------------------
// p_bar = mean over P of p  ([P,B,D] -> [B,D])
// ---------------------------------------------------------------------------
__global__ void pbar_kernel(const float* __restrict__ p,
                            float* __restrict__ pbar) {
  int i = blockIdx.x * 256 + threadIdx.x;   // over B*D
  float s = 0.f;
#pragma unroll
  for (int l = 0; l < P_SZ; ++l)
    s += p[(size_t)l * B_SZ * DM + i];
  pbar[i] = s * (1.f / P_SZ);
}

// ---------------------------------------------------------------------------
// Symmetric KL over a 128-row held 4 elems/lane by one wave32.
// ---------------------------------------------------------------------------
__device__ __forceinline__ float wred_max(float v) {
#pragma unroll
  for (int o = 16; o; o >>= 1) v = fmaxf(v, __shfl_xor(v, o, 32));
  return v;
}
__device__ __forceinline__ float wred_sum(float v) {
#pragma unroll
  for (int o = 16; o; o >>= 1) v += __shfl_xor(v, o, 32);
  return v;
}

__device__ __forceinline__ float sym_kl4(float4 ys, float4 yt) {
  const float invT = 0.25f;                       // 1/T, T=4
  float a[4] = {ys.x * invT, ys.y * invT, ys.z * invT, ys.w * invT};
  float b[4] = {yt.x * invT, yt.y * invT, yt.z * invT, yt.w * invT};
  float ma = wred_max(fmaxf(fmaxf(a[0], a[1]), fmaxf(a[2], a[3])));
  float mb = wred_max(fmaxf(fmaxf(b[0], b[1]), fmaxf(b[2], b[3])));
  float ea4[4], eb4[4];
  float ea = 0.f, eb = 0.f;
#pragma unroll
  for (int j = 0; j < 4; ++j) {
    ea4[j] = __expf(a[j] - ma);
    eb4[j] = __expf(b[j] - mb);
    ea += ea4[j];
    eb += eb4[j];
  }
  ea = wred_sum(ea);
  eb = wred_sum(eb);
  float ra = __builtin_amdgcn_rcpf(ea);   // softmax probs = e_j * rcp(sum)
  float rb = __builtin_amdgcn_rcpf(eb);
  float la = ma + __logf(ea);
  float lb = mb + __logf(eb);
  float s = 0.f;
#pragma unroll
  for (int j = 0; j < 4; ++j) {
    float ls = a[j] - la;                 // log_softmax(ys/T)
    float lt = b[j] - lb;                 // log_softmax(yt/T)
    s += (lt - ls) * (eb4[j] * rb - ea4[j] * ra);  // pt*(lt-ls)+ps*(ls-lt)
  }
  s = wred_sum(s);
  return s * (16.f / 128.f);   // * T^2 / D
}

__global__ void dil_kernel(const float* __restrict__ g,
                           const float* __restrict__ pbar,
                           float* __restrict__ out) {
  int wid  = (blockIdx.x * blockDim.x + threadIdx.x) >> 5;
  int lane = threadIdx.x & 31;
  float4 ys = *(const float4*)&g[(size_t)wid * DM + lane * 4];
  float4 yt = *(const float4*)&pbar[(size_t)wid * DM + lane * 4];
  float v = sym_kl4(ys, yt);
  if (lane == 0) atomicAdd(out, v);
}

__global__ void dcl_kernel(const float* __restrict__ g,
                           const float* __restrict__ pbar,
                           const float* __restrict__ p,
                           float* __restrict__ out) {
  int rid  = (blockIdx.x * blockDim.x + threadIdx.x) >> 5;   // l*B + b
  int lane = threadIdx.x & 31;
  int b = rid & (B_SZ - 1);
  float4 gg = *(const float4*)&g[(size_t)b * DM + lane * 4];
  float4 pb = *(const float4*)&pbar[(size_t)b * DM + lane * 4];
  float4 pp = *(const float4*)&p[(size_t)rid * DM + lane * 4];
  float4 d1, d2;
  d1.x = (gg.x - pp.x) * (gg.x - pp.x);
  d1.y = (gg.y - pp.y) * (gg.y - pp.y);
  d1.z = (gg.z - pp.z) * (gg.z - pp.z);
  d1.w = (gg.w - pp.w) * (gg.w - pp.w);
  d2.x = (pb.x - pp.x) * (pb.x - pp.x);
  d2.y = (pb.y - pp.y) * (pb.y - pp.y);
  d2.z = (pb.z - pp.z) * (pb.z - pp.z);
  d2.w = (pb.w - pp.w) * (pb.w - pp.w);
  float v = sym_kl4(d1, d2) * (1.f / P_SZ);
  if (lane == 0) atomicAdd(out + 1, v);
}

__global__ void zero_out_kernel(float* out) {
  if (threadIdx.x < 2) out[threadIdx.x] = 0.f;
}

// ---------------------------------------------------------------------------
extern "C" void kernel_launch(void* const* d_in, const int* in_sizes, int n_in,
                              void* d_out, int out_size, void* d_ws,
                              size_t ws_size, hipStream_t stream) {
  const float* ebg    = (const float*)d_in[0];
  const float* ebp    = (const float*)d_in[1];
  // d_in[2] = labelsg (unused by reference outputs)
  const float* glo_w1 = (const float*)d_in[3];
  const float* glo_b1 = (const float*)d_in[4];
  const float* glo_w2 = (const float*)d_in[5];
  const float* glo_b2 = (const float*)d_in[6];
  const float* pat_w1 = (const float*)d_in[7];
  const float* pat_b1 = (const float*)d_in[8];
  const float* pat_w2 = (const float*)d_in[9];
  const float* pat_b2 = (const float*)d_in[10];
  float* out = (float*)d_out;

  char* ws = (char*)d_ws;
  const size_t W1P_ELEMS = (size_t)8 * KS1 * 32 * 16;   // 81920 bf16
  const size_t W2P_ELEMS = (size_t)8 * KS2 * 32 * 16;   // 16384 bf16
  unsigned short* w1g = (unsigned short*)ws; ws += W1P_ELEMS * 2;
  unsigned short* w2g = (unsigned short*)ws; ws += W2P_ELEMS * 2;
  unsigned short* w1p = (unsigned short*)ws; ws += W1P_ELEMS * 2;
  unsigned short* w2p = (unsigned short*)ws; ws += W2P_ELEMS * 2;
  float* gbuf    = (float*)ws; ws += (size_t)B_SZ * DM * 4;
  float* pbuf    = (float*)ws; ws += (size_t)P_SZ * B_SZ * DM * 4;
  float* pbarbuf = (float*)ws;

  zero_out_kernel<<<1, 32, 0, stream>>>(out);

  pack_w_kernel<<<(int)((W1P_ELEMS + 255) / 256), 256, 0, stream>>>(glo_w1, w1g, DIN);
  pack_w_kernel<<<(int)((W2P_ELEMS + 255) / 256), 256, 0, stream>>>(glo_w2, w2g, DM);
  pack_w_kernel<<<(int)((W1P_ELEMS + 255) / 256), 256, 0, stream>>>(pat_w1, w1p, DIN);
  pack_w_kernel<<<(int)((W2P_ELEMS + 255) / 256), 256, 0, stream>>>(pat_w2, w2p, DM);

  embed_kernel<<<B_SZ / MT, 256, 0, stream>>>(ebg, w1g, glo_b1, w2g, glo_b2, gbuf);
  embed_kernel<<<(P_SZ * B_SZ) / MT, 256, 0, stream>>>(ebp, w1p, pat_b1, w2p, pat_b2, pbuf);

  pbar_kernel<<<(B_SZ * DM) / 256, 256, 0, stream>>>(pbuf, pbarbuf);
  dil_kernel<<<(B_SZ * 32) / 256, 256, 0, stream>>>(gbuf, pbarbuf, out);
  dcl_kernel<<<(P_SZ * B_SZ * 32) / 256, 256, 0, stream>>>(gbuf, pbarbuf, pbuf, out);
}
